// MultiHeadAttention_80650895884552
// MI455X (gfx1250) — compile-verified, matrix operand order checked by reading, not run on hardware
//
#include <hip/hip_runtime.h>

// Problem constants (from reference): B=4, S=2048, D=1024, H=16, DK=DV=64
#define BB  4
#define SS  2048
#define DDIM 1024
#define HH  16
#define DKV 64

typedef __attribute__((ext_vector_type(16))) __bf16 v16bf;
typedef __attribute__((ext_vector_type(8)))  __bf16 v8bf;
typedef __attribute__((ext_vector_type(8)))  float  v8f;

// CDNA5 16-bit A/B fragment element->K mapping (ISA 05_wmma 7.12.2):
// lane half 0: e0..7 -> K=0..7,  e8..15 -> K=16..23
// lane half 1: e0..7 -> K=8..15, e8..15 -> K=24..31

// Fragment load from f32 memory; lane-index dim slow, K contiguous.
// (used only for the raw f32 `tokens` input)
__device__ __forceinline__ v16bf load_frag_f32(const float* __restrict__ src,
                                               int ld, int lane) {
    int half = lane >> 4, r = lane & 15;
    const float* p = src + (size_t)r * ld + 8 * half;
    float4 x0 = ((const float4*)p)[0];        // K = 8h + 0..3
    float4 x1 = ((const float4*)p)[1];        // K = 8h + 4..7
    float4 x2 = ((const float4*)(p + 16))[0]; // K = 16 + 8h + 0..3
    float4 x3 = ((const float4*)(p + 16))[1]; // K = 16 + 8h + 4..7
    v16bf f;
    f[0]  = (__bf16)x0.x; f[1]  = (__bf16)x0.y; f[2]  = (__bf16)x0.z; f[3]  = (__bf16)x0.w;
    f[4]  = (__bf16)x1.x; f[5]  = (__bf16)x1.y; f[6]  = (__bf16)x1.z; f[7]  = (__bf16)x1.w;
    f[8]  = (__bf16)x2.x; f[9]  = (__bf16)x2.y; f[10] = (__bf16)x2.z; f[11] = (__bf16)x2.w;
    f[12] = (__bf16)x3.x; f[13] = (__bf16)x3.y; f[14] = (__bf16)x3.z; f[15] = (__bf16)x3.w;
    return f;
}

// Fragment load from bf16 memory; lane-index dim slow, K contiguous.
// Per lane: two contiguous 8-element (16-byte) runs -> 2 x global_load_b128.
__device__ __forceinline__ v16bf load_frag_bf16(const __bf16* __restrict__ src,
                                                int ld, int lane) {
    int half = lane >> 4, r = lane & 15;
    const __bf16* p = src + (size_t)r * ld + 8 * half;
    v8bf lo = *(const v8bf*)p;          // e0..7 : K = 8h .. 8h+7
    v8bf hi = *(const v8bf*)(p + 16);   // e8..15: K = 16+8h .. 23+8h
    v16bf f;
#pragma unroll
    for (int e = 0; e < 8; ++e) { f[e] = lo[e]; f[e + 8] = hi[e]; }
    return f;
}

__device__ __forceinline__ v8f wmma_bf16(v16bf a, v16bf b, v8f c) {
    return __builtin_amdgcn_wmma_f32_16x16x32_bf16(
        false, a, false, b, (short)0, c, false, false);
}

// ---------------------------------------------------------------------------
// Kernel 0: one-time weight transpose + f32->bf16 downconvert.
// W: [Hn][K][N] f32  ->  Wt: [Hn][N][K] bf16   (N-major, so B fragments are
// two contiguous b128 runs per lane). Reads coalesce across threads (n);
// writes are per-thread contiguous 16B chunks.
// ---------------------------------------------------------------------------
__global__ void wtrans_kernel(const float* __restrict__ W, __bf16* __restrict__ Wt,
                              int K, int N) {
    int h = blockIdx.y;
    int n = blockIdx.x * blockDim.x + threadIdx.x;
    if (n >= N) return;
    const float* src = W + (size_t)h * K * N + n;
    __bf16* dst = Wt + ((size_t)h * N + n) * K;
    for (int d = 0; d < K; d += 8) {
        v8bf t;
#pragma unroll
        for (int e = 0; e < 8; ++e) t[e] = (__bf16)src[(size_t)(d + e) * N];
        *(v8bf*)(dst + d) = t;
    }
}

// ---------------------------------------------------------------------------
// Kernel 1: fused QKV projection.  One wave -> 16 token rows x 64 cols for one
// head; Q, K, V share the A fragment each k-step (12 WMMAs / k-step).
// Outputs bf16: Q,K as [B,H,S,64]; V transposed as [B,H,64,S] so the PV
// B-fragments in kernel 2 are contiguous loads.
// ---------------------------------------------------------------------------
__global__ __launch_bounds__(128, 1)
void qkv_kernel(const float* __restrict__ X,
                const __bf16* __restrict__ WqT, const __bf16* __restrict__ WkT,
                const __bf16* __restrict__ WvT,
                __bf16* __restrict__ Q, __bf16* __restrict__ K,
                __bf16* __restrict__ VT) {
    int lane = threadIdx.x & 31, wave = threadIdx.x >> 5;
    int h    = blockIdx.x % HH;
    int row0 = (blockIdx.x / HH) * 64 + wave * 16;   // flat token row [0, B*S)
    const __bf16* wq = WqT + (size_t)h * DKV * DDIM; // [n][d] bf16
    const __bf16* wk = WkT + (size_t)h * DKV * DDIM;
    const __bf16* wv = WvT + (size_t)h * DKV * DDIM;

    v8f cq[4] = {}, ck[4] = {}, cv[4] = {};
    for (int k0 = 0; k0 < DDIM; k0 += 32) {
        v16bf a = load_frag_f32(X + (size_t)row0 * DDIM + k0, DDIM, lane);
#pragma unroll
        for (int f = 0; f < 4; ++f) {
            cq[f] = wmma_bf16(a, load_frag_bf16(wq + (size_t)(f * 16) * DDIM + k0, DDIM, lane), cq[f]);
            ck[f] = wmma_bf16(a, load_frag_bf16(wk + (size_t)(f * 16) * DDIM + k0, DDIM, lane), ck[f]);
            cv[f] = wmma_bf16(a, load_frag_bf16(wv + (size_t)(f * 16) * DDIM + k0, DDIM, lane), cv[f]);
        }
    }
    int half = lane >> 4, nl = lane & 15;
    int bh0 = (row0 >> 11) * HH + h;                  // (b, h) flat
    int s0  = row0 & (SS - 1);
#pragma unroll
    for (int f = 0; f < 4; ++f)
#pragma unroll
        for (int r = 0; r < 8; ++r) {
            int s = s0 + r + 8 * half;
            int d = f * 16 + nl;
            size_t idx = ((size_t)bh0 * SS + s) * DKV + d;            // [bh][s][d]
            Q[idx] = (__bf16)cq[f][r];
            K[idx] = (__bf16)ck[f][r];
            size_t tdx = ((size_t)bh0 * DKV + d) * SS + s;            // [bh][d][s]
            VT[tdx] = (__bf16)cv[f][r];
        }
}

// ---------------------------------------------------------------------------
// Kernel 2: flash attention with the reference's swapped softmax:
//   heads[j,v] = sum_i softmax_i(k_j . q_i / 8) * V[i,v]
// Scores computed transposed (M=i, N=j) so the softmax axis reduces in-lane +
// one shfl_xor(16); the f32 C tiles of T map lane-for-lane onto the bf16 A
// fragment of P^T for the PV WMMA (zero cross-lane movement).
// ---------------------------------------------------------------------------
__global__ __launch_bounds__(128, 1)
void attn_kernel(const __bf16* __restrict__ Q, const __bf16* __restrict__ K,
                 const __bf16* __restrict__ VT, __bf16* __restrict__ M) {
    int lane = threadIdx.x & 31, wave = threadIdx.x >> 5;
    int jt = blockIdx.x % (SS / 64);
    int bh = blockIdx.x / (SS / 64);
    int b = bh / HH, h = bh % HH;
    const __bf16* Qh  = Q  + (size_t)bh * SS * DKV;
    const __bf16* Kh  = K  + (size_t)bh * SS * DKV;
    const __bf16* VTh = VT + (size_t)bh * DKV * SS;   // [v][s]
    int j0   = jt * 64 + wave * 16;
    int half = lane >> 4, nl = lane & 15;

    // B = K^T for this wave's 16 j-columns (lane = j, contiguous along d)
    v16bf bk0 = load_frag_bf16(Kh + (size_t)j0 * DKV +  0, DKV, lane);
    v16bf bk1 = load_frag_bf16(Kh + (size_t)j0 * DKV + 32, DKV, lane);

    const float SC = 0.125f * 1.4426950408889634f;  // 1/sqrt(64) * log2(e)
    float mrun = -3.0e38f, srun = 0.f;
    v8f o[4] = {};

    for (int i0 = 0; i0 < SS; i0 += 32) {
        v16bf a00 = load_frag_bf16(Qh + (size_t)i0 * DKV,             DKV, lane);
        v16bf a01 = load_frag_bf16(Qh + (size_t)i0 * DKV + 32,        DKV, lane);
        v16bf a10 = load_frag_bf16(Qh + (size_t)(i0 + 16) * DKV,      DKV, lane);
        v16bf a11 = load_frag_bf16(Qh + (size_t)(i0 + 16) * DKV + 32, DKV, lane);
        v8f z = {};
        v8f t0 = wmma_bf16(a01, bk1, wmma_bf16(a00, bk0, z));  // i = i0..i0+15
        v8f t1 = wmma_bf16(a11, bk1, wmma_bf16(a10, bk0, z));  // i = i0+16..+31

        float tmax = -3.0e38f;
#pragma unroll
        for (int r = 0; r < 8; ++r) {
            t0[r] *= SC; t1[r] *= SC;
            tmax = fmaxf(tmax, fmaxf(t0[r], t1[r]));
        }
        tmax = fmaxf(tmax, __shfl_xor(tmax, 16, 32));
        float mnew = fmaxf(mrun, tmax);
        float corr = __builtin_amdgcn_exp2f(mrun - mnew);

        // exp + pack probs straight into the P^T A-fragment (lane row = j)
        v16bf ap;
        float psum = 0.f;
#pragma unroll
        for (int r = 0; r < 8; ++r) {
            float p0 = __builtin_amdgcn_exp2f(t0[r] - mnew);
            float p1 = __builtin_amdgcn_exp2f(t1[r] - mnew);
            psum += p0 + p1;
            ap[r]     = (__bf16)p0;   // K = r + 8*half      (i tile 0)
            ap[r + 8] = (__bf16)p1;   // K = r + 16 + 8*half (i tile 1)
        }
        psum += __shfl_xor(psum, 16, 32);
        srun = srun * corr + psum;
        mrun = mnew;

        // Rescale running O: row j of O = r + 8*half; stats live at lane j.
#pragma unroll
        for (int r = 0; r < 8; ++r) {
            float cj = __shfl(corr, r + 8 * half, 32);
#pragma unroll
            for (int f = 0; f < 4; ++f) o[f][r] *= cj;
        }
        // O[j,v] += P^T @ V  (K = 32 i-values); B from transposed V: [v][i]
#pragma unroll
        for (int f = 0; f < 4; ++f)
            o[f] = wmma_bf16(ap, load_frag_bf16(VTh + (size_t)(f * 16) * SS + i0, SS, lane), o[f]);
    }

    float inv = 1.0f / srun;
#pragma unroll
    for (int r = 0; r < 8; ++r) {
        float ir = __shfl(inv, r + 8 * half, 32);
        int j = j0 + r + 8 * half;
        size_t base = ((size_t)b * SS + j) * (HH * DKV) + h * DKV;  // multi[b, j, h*64+v]
#pragma unroll
        for (int f = 0; f < 4; ++f) M[base + f * 16 + nl] = (__bf16)(o[f][r] * ir);
    }
}

// ---------------------------------------------------------------------------
// Kernel 3: out-projection + residual + LayerNorm, fused. One workgroup owns
// 16 full rows (8 waves x 128 cols). Row stats: in-lane over fragments ->
// shfl_xor tree over 16 lanes -> tiny LDS cross-wave combine. The C fragment
// row mapping (M = r + 8*half) matches the stats layout, so normalization is
// applied per-register with no shuffles.
// ---------------------------------------------------------------------------
__global__ __launch_bounds__(256, 1)
void proj_ln_kernel(const __bf16* __restrict__ Mi, const __bf16* __restrict__ WoT,
                    const float* __restrict__ X, const float* __restrict__ gamma,
                    const float* __restrict__ beta, float* __restrict__ out) {
    __shared__ float ls[8][16], lq[8][16];
    int lane = threadIdx.x & 31, wave = threadIdx.x >> 5;
    int row0 = blockIdx.x * 16;
    int col0 = wave * 128;

    v8f c[8] = {};
    for (int k0 = 0; k0 < DDIM; k0 += 32) {
        v16bf a = load_frag_bf16(Mi + (size_t)row0 * DDIM + k0, DDIM, lane);
#pragma unroll
        for (int f = 0; f < 8; ++f)
            c[f] = wmma_bf16(a, load_frag_bf16(WoT + (size_t)(col0 + f * 16) * DDIM + k0, DDIM, lane), c[f]);
    }

    int half = lane >> 4, nl = lane & 15;
    float ps[8], pq[8];
#pragma unroll
    for (int r = 0; r < 8; ++r) { ps[r] = 0.f; pq[r] = 0.f; }
#pragma unroll
    for (int f = 0; f < 8; ++f) {
        int col = col0 + f * 16 + nl;
#pragma unroll
        for (int r = 0; r < 8; ++r) {
            int row = row0 + r + 8 * half;
            float x = c[f][r] + X[(size_t)row * DDIM + col];   // residual
            c[f][r] = x;
            ps[r] += x;
            pq[r] += x * x;
        }
    }
#pragma unroll
    for (int r = 0; r < 8; ++r) {
#pragma unroll
        for (int d = 1; d < 16; d <<= 1) {
            ps[r] += __shfl_xor(ps[r], d, 32);
            pq[r] += __shfl_xor(pq[r], d, 32);
        }
    }
    if (nl == 0) {
#pragma unroll
        for (int r = 0; r < 8; ++r) {
            ls[wave][r + 8 * half] = ps[r];
            lq[wave][r + 8 * half] = pq[r];
        }
    }
    __syncthreads();

#pragma unroll
    for (int r = 0; r < 8; ++r) {
        int j = r + 8 * half;
        float S = 0.f, Qs = 0.f;
#pragma unroll
        for (int w = 0; w < 8; ++w) { S += ls[w][j]; Qs += lq[w][j]; }
        float mu   = S * (1.0f / DDIM);
        float var  = Qs * (1.0f / DDIM) - mu * mu;
        float rstd = __frsqrt_rn(var + 1e-5f);
        int row = row0 + j;
#pragma unroll
        for (int f = 0; f < 8; ++f) {
            int col = col0 + f * 16 + nl;
            out[(size_t)row * DDIM + col] = (c[f][r] - mu) * rstd * gamma[col] + beta[col];
        }
    }
}

extern "C" void kernel_launch(void* const* d_in, const int* in_sizes, int n_in,
                              void* d_out, int out_size, void* d_ws, size_t ws_size,
                              hipStream_t stream) {
    const float* tokens = (const float*)d_in[0];
    const float* Wq     = (const float*)d_in[1];
    const float* Wk     = (const float*)d_in[2];
    const float* Wv     = (const float*)d_in[3];
    const float* Wo     = (const float*)d_in[4];
    const float* gamma  = (const float*)d_in[5];
    const float* beta   = (const float*)d_in[6];
    float* out = (float*)d_out;

    // Workspace layout (bf16 elements):
    const size_t wsz  = (size_t)HH * DDIM * DKV;     // 1,048,576 (2 MiB)
    const size_t tsz  = (size_t)BB * HH * SS * DKV;  // 8,388,608 (16 MiB)
    __bf16* wqT = (__bf16*)d_ws;
    __bf16* wkT = wqT + wsz;
    __bf16* wvT = wkT + wsz;
    __bf16* woT = wvT + wsz;                         // 1024x1024
    __bf16* qb  = woT + (size_t)DDIM * DDIM;
    __bf16* kb  = qb + tsz;
    __bf16* vtb = kb + tsz;                          // [bh][v][s]
    __bf16* mb  = vtb + tsz;                         // [B*S][H*DV]

    // One-time weight transpose + downconvert (f32 [h][d][n] -> bf16 [h][n][d])
    wtrans_kernel<<<dim3(1, HH), dim3(DKV), 0, stream>>>(Wq, wqT, DDIM, DKV);
    wtrans_kernel<<<dim3(1, HH), dim3(DKV), 0, stream>>>(Wk, wkT, DDIM, DKV);
    wtrans_kernel<<<dim3(1, HH), dim3(DKV), 0, stream>>>(Wv, wvT, DDIM, DKV);
    wtrans_kernel<<<dim3(DDIM / 256, 1), dim3(256), 0, stream>>>(Wo, woT, DDIM, DDIM);

    qkv_kernel<<<dim3((BB * SS / 64) * HH), dim3(128), 0, stream>>>(
        tokens, wqT, wkT, wvT, qb, kb, vtb);
    attn_kernel<<<dim3(BB * HH * (SS / 64)), dim3(128), 0, stream>>>(
        qb, kb, vtb, mb);
    proj_ln_kernel<<<dim3(BB * SS / 16), dim3(256), 0, stream>>>(
        mb, woT, tokens, gamma, beta, out);
}
